// Gtu_39230231282335
// MI455X (gfx1250) — matrix operand order repaired and checked by reading, not compiled, not run on hardware
//
#include <hip/hip_runtime.h>
#include <hip/hip_bf16.h>
#include <stdint.h>

typedef _Float16 f16;
typedef __attribute__((ext_vector_type(16))) _Float16 v16h;
typedef __attribute__((ext_vector_type(8)))  _Float16 v8h;
typedef __attribute__((ext_vector_type(8)))  float    v8f;

struct __attribute__((packed, aligned(2))) h8u { f16 h[8]; };

#define NH   8
#define EDIM 512
#define D1   1536
#define BB   4
#define NN   2048
#define M2   4096   // 2n
#define RPED 512

// ---------------- f32 -> f16 convert ----------------
__global__ void cvt_f16_kernel(const float* __restrict__ in, f16* __restrict__ out, int n) {
  int i = blockIdx.x * blockDim.x + threadIdx.x;
  if (i < n) out[i] = (f16)in[i];
}

// ---------------- RPE input layer: h0[m,j] = idx(m)*Wp[j] + bp[j] ----------------
__global__ void rpe_h0_kernel(const float* __restrict__ Wp, const float* __restrict__ bp,
                              float* __restrict__ h0) {
  int i = blockIdx.x * blockDim.x + threadIdx.x;
  if (i >= M2 * RPED) return;
  int m = i >> 9, j = i & 511;
  float idx;
  if (m == 0 || m == NN)      idx = 0.f;
  else if (m < NN)            idx = (float)m;
  else                        idx = (float)(m - M2);
  h0[i] = idx * Wp[j] + bp[j];
}

// ---------------- SimpleRMSNorm + ReLU + cast to f16 (rows x 512) ----------------
__global__ void srms_relu_f16_kernel(const float* __restrict__ in, f16* __restrict__ out, int rows) {
  int wave = (blockIdx.x * blockDim.x + threadIdx.x) >> 5;
  int lane = threadIdx.x & 31;
  if (wave >= rows) return;
  const float* r = in + (size_t)wave * RPED;
  float x[16];
  float ss = 0.f;
#pragma unroll
  for (int t = 0; t < 16; ++t) { x[t] = r[lane + 32 * t]; ss += x[t] * x[t]; }
#pragma unroll
  for (int off = 16; off; off >>= 1) ss += __shfl_xor(ss, off, 32);
  float rms = sqrtf(ss) * 0.04419417382415922f;   // * 512^-0.5
  float inv = 1.f / (rms + 1e-8f);
  f16* o = out + (size_t)wave * RPED;
#pragma unroll
  for (int t = 0; t < 16; ++t) {
    float y = x[t] * inv;
    y = y > 0.f ? y : 0.f;
    o[lane + 32 * t] = (f16)y;
  }
}

// ---------------- generic WMMA GEMM:  C(MxN) = A(MxK,f16) @ W(NxK,f16)^T + bias ----------------
// MODE 0: out f32 row-major (+bias)
// MODE 1: SiLU, out f32 row-major               (u projection)
// MODE 2: SiLU, out f16 transposed (b,c,s)      (v projection -> Vt)
// MODE 3: out f16 scattered reversed-extended   (RPE final -> AR[c][ (2047-m) & 4095 ])
template <int MODE>
__global__ void wmma_gemm_kernel(const f16* __restrict__ A, const f16* __restrict__ W,
                                 const float* __restrict__ bias,
                                 float* __restrict__ outF, f16* __restrict__ outH,
                                 int M, int N, int K) {
  int wid  = (blockIdx.x * blockDim.x + threadIdx.x) >> 5;
  int lane = threadIdx.x & 31;
  int nstrips = N >> 6;                       // 64-wide N strips
  int mtile = wid / nstrips;
  int nstrip = wid - mtile * nstrips;
  if (mtile >= (M >> 4)) return;
  int lo = lane & 15, hi = lane >> 4;
  int row = (mtile << 4) + lo;
  const f16* Arow = A + (size_t)row * K;
  const f16* Wcol = W + (size_t)((nstrip << 6) + lo) * K + 16 * hi;  // per-lane col base

  v8f acc[4] = {};
  for (int k0 = 0; k0 < K; k0 += 32) {
    // ---- batch ALL loads first (one clause, one wait), then 4 back-to-back WMMAs
    v8h  r0 = *(const v8h*)(Arow + k0 + 16 * hi);       // A halves 0..7  = K 16*hi..+7
    v8h  r1 = *(const v8h*)(Arow + k0 + 16 * hi + 8);   // A halves 8..15 = K 16*hi+8..+15
    v16h bf[4];
#pragma unroll
    for (int t = 0; t < 4; ++t)
      bf[t] = *(const v16h*)(Wcol + (size_t)(t << 4) * K + k0);      // col N strip t, K 16*hi..+15
    v16h af;
#pragma unroll
    for (int i = 0; i < 8; ++i) { af[i] = r0[i]; af[8 + i] = r1[i]; }
#pragma unroll
    for (int t = 0; t < 4; ++t)
      acc[t] = __builtin_amdgcn_wmma_f32_16x16x32_f16(false, af, false, bf[t],
                                                      (short)0, acc[t], false, false);
  }

#pragma unroll
  for (int t = 0; t < 4; ++t) {
    int col = (nstrip << 6) + (t << 4) + lo;
    float bv = bias[col];
#pragma unroll
    for (int r = 0; r < 8; ++r) {
      int m = (mtile << 4) + r + 8 * hi;      // D layout: VGPR r -> M = r + 8*(lane/16)
      float y = acc[t][r] + bv;
      if (MODE == 1 || MODE == 2) y = y / (1.f + __expf(-y));   // SiLU
      if (MODE == 0 || MODE == 1) {
        outF[(size_t)m * N + col] = y;
      } else if (MODE == 2) {
        int b = m >> 11, s = m & 2047;        // m = b*2048 + s
        outH[(((size_t)(b * D1 + col)) << 11) + s] = (f16)y;    // Vt[b][c][s]
      } else {                                 // MODE 3: AR[c][k], k = (2047-m) mod 4096
        outH[((size_t)col << 12) + ((2047 - m) & 4095)] = (f16)y;
      }
    }
  }
}

// ---------------- per-channel Toeplitz conv via WMMA + gate epilogue ----------------
// out[b,t,c] = sum_s Vt[b][c][s] * a_c[(t-s) mod 4096];  P = f16( U * out )
// A-frag element (t-row, K=s) = AR[c][s + 2047 - t]  (ascending contiguous)
__global__ void toeplitz_conv_kernel(const f16* __restrict__ AR, const f16* __restrict__ Vt,
                                     const float* __restrict__ U, f16* __restrict__ P) {
  int wid  = (blockIdx.x * blockDim.x + threadIdx.x) >> 5;
  int lane = threadIdx.x & 31;
  int c  = wid >> 5;                 // channel (h*192 + d), 1536 total
  int tg = wid & 31;                 // 32 groups of 64 t-rows
  if (c >= D1) return;
  int lo = lane & 15, hi = lane >> 4;
  const f16* ar = AR + ((size_t)c << 12);
  bool haveB = lo < BB;
  const f16* vrow = Vt + (((size_t)(lo * D1 + c)) << 11) + 16 * hi;  // only read when haveB
  int tbase = tg << 6;

  // B fragment: zero-init ONCE; padded N columns (lo>=4) keep zero forever.
  v16h bf;
#pragma unroll
  for (int i = 0; i < 16; ++i) bf[i] = (f16)0.f;

  v8f acc[4] = {};
  for (int s0 = 0; s0 < NN; s0 += 32) {
    if (haveB) bf = *(const v16h*)(vrow + s0);           // col N = batch lo, K 16*hi..+15
    // ---- batch all 8 A-fragment b128 loads, then 4 back-to-back WMMAs
    v16h af[4];
#pragma unroll
    for (int t4 = 0; t4 < 4; ++t4) {
      int trow = tbase + (t4 << 4) + lo;
      int base0 = s0 + 16 * hi + 2047 - trow;            // in [0, 4094-15]
      h8u r0 = *(const h8u*)(ar + base0);                // unaligned 8-half run (K run0)
      h8u r1 = *(const h8u*)(ar + base0 + 8);            // K run1
#pragma unroll
      for (int i = 0; i < 8; ++i) { af[t4][i] = r0.h[i]; af[t4][8 + i] = r1.h[i]; }
    }
#pragma unroll
    for (int t4 = 0; t4 < 4; ++t4)
      acc[t4] = __builtin_amdgcn_wmma_f32_16x16x32_f16(false, af[t4], false, bf,
                                                       (short)0, acc[t4], false, false);
  }

  if (haveB) {                                           // lanes with N-col = batch 0..3
    int b = lo;
#pragma unroll
    for (int t4 = 0; t4 < 4; ++t4)
#pragma unroll
      for (int r = 0; r < 8; ++r) {
        int t = tbase + (t4 << 4) + r + 8 * hi;
        size_t oi = ((size_t)(b * NN + t)) * D1 + c;
        P[oi] = (f16)(U[oi] * acc[t4][r]);               // gate with u (SiLU already applied)
      }
  }
}

// ---------------- host launch ----------------
static inline int gemm_blocks(int M, int N) { return ((M >> 4) * (N >> 6) + 7) / 8; }

extern "C" void kernel_launch(void* const* d_in, const int* in_sizes, int n_in,
                              void* d_out, int out_size, void* d_ws, size_t ws_size,
                              hipStream_t stream) {
  const float* x  = (const float*)d_in[0];
  const float* Wu = (const float*)d_in[1];
  const float* bu = (const float*)d_in[2];
  const float* Wv = (const float*)d_in[3];
  const float* bv = (const float*)d_in[4];
  const float* Wo = (const float*)d_in[5];
  const float* bo = (const float*)d_in[6];
  const float* Wp = (const float*)d_in[7];
  const float* bp = (const float*)d_in[8];
  const float* Wl = (const float*)d_in[9];
  const float* bl = (const float*)d_in[10];
  const float* Wr = (const float*)d_in[11];
  const float* br = (const float*)d_in[12];
  float* out = (float*)d_out;

  char* w = (char*)d_ws;
  f16*   xh  = (f16*)  (w);               // 8,388,608  (8192x512 f16)
  float* h0  = (float*)(w + 8388608);     // 8,388,608  (4096x512 f32)
  float* h1  = (float*)(w + 16777216);    // 8,388,608
  f16*   P   = (f16*)  (w);               // 25,165,824 overlays xh/h0/h1 (dead by then)
  f16*   Wuh = (f16*)  (w + 25165824);    // 1,572,864
  f16*   Wvh = (f16*)  (w + 26738688);    // 1,572,864
  f16*   Woh = (f16*)  (w + 28311552);    // 1,572,864
  f16*   Wlh = (f16*)  (w + 29884416);    // 1,572,864 (3x512x512)
  f16*   Wrh = (f16*)  (w + 31457280);    // 1,572,864
  f16*   hn  = (f16*)  (w + 33030144);    // 4,194,304 (4096x512 f16)
  float* U   = (float*)(w + 37224448);    // 50,331,648 (8192x1536 f32)
  f16*   Vt  = (f16*)  (w + 87556096);    // 25,165,824 (4x1536x2048 f16)
  f16*   ARv = (f16*)  (w + 112721920);   // 12,582,912 (1536x4096 f16)  total ~119.5 MB

  // f32 -> f16 weight/activation casts
  cvt_f16_kernel<<<(4194304 + 255) / 256, 256, 0, stream>>>(x,  xh,  4194304);
  cvt_f16_kernel<<<(786432 + 255) / 256, 256, 0, stream>>>(Wu, Wuh, 786432);
  cvt_f16_kernel<<<(786432 + 255) / 256, 256, 0, stream>>>(Wv, Wvh, 786432);
  cvt_f16_kernel<<<(786432 + 255) / 256, 256, 0, stream>>>(Wo, Woh, 786432);
  cvt_f16_kernel<<<(786432 + 255) / 256, 256, 0, stream>>>(Wl, Wlh, 786432);
  cvt_f16_kernel<<<(786432 + 255) / 256, 256, 0, stream>>>(Wr, Wrh, 786432);

  // RPE net: h0 = idx*Wp + bp ; 3x (srms->relu->f16 -> GEMM 512x512 + bl)
  rpe_h0_kernel<<<(M2 * RPED + 255) / 256, 256, 0, stream>>>(Wp, bp, h0);
  float* hc = h0; float* hx = h1;
  for (int i = 0; i < 3; ++i) {
    srms_relu_f16_kernel<<<(M2 + 7) / 8, 256, 0, stream>>>(hc, hn, M2);
    wmma_gemm_kernel<0><<<gemm_blocks(M2, 512), 256, 0, stream>>>(
        hn, Wlh + (size_t)i * 262144, bl + (size_t)i * 512, hx, nullptr, M2, 512, 512);
    float* tmp = hc; hc = hx; hx = tmp;
  }
  // final RPE layer -> scatter reversed-extended kernel AR[c][k]
  srms_relu_f16_kernel<<<(M2 + 7) / 8, 256, 0, stream>>>(hc, hn, M2);
  wmma_gemm_kernel<3><<<gemm_blocks(M2, D1), 256, 0, stream>>>(
      hn, Wrh, br, nullptr, ARv, M2, D1, 512);

  // u = silu(x@Wu^T+bu) (f32 row-major); v = silu(x@Wv^T+bv) transposed -> Vt[b][c][s] f16
  wmma_gemm_kernel<1><<<gemm_blocks(BB * NN, D1), 256, 0, stream>>>(
      xh, Wuh, bu, U, nullptr, BB * NN, D1, EDIM);
  wmma_gemm_kernel<2><<<gemm_blocks(BB * NN, D1), 256, 0, stream>>>(
      xh, Wvh, bv, nullptr, Vt, BB * NN, D1, EDIM);

  // per-channel Toeplitz conv + gate:  P = f16(U * (T_c @ v))  (1536 channels x 32 t-groups)
  toeplitz_conv_kernel<<<(D1 * 32) / 8, 256, 0, stream>>>(ARv, Vt, U, P);

  // out = P @ Wo^T + bo  (f32)
  wmma_gemm_kernel<0><<<gemm_blocks(BB * NN, EDIM), 256, 0, stream>>>(
      P, Woh, bo, out, nullptr, BB * NN, EDIM, D1);
}
